// GAT_38878043963727
// MI455X (gfx1250) — matrix operand order repaired and checked by reading, not compiled, not run on hardware
//
#include <hip/hip_runtime.h>

typedef __attribute__((ext_vector_type(16))) _Float16 v16h;
typedef __attribute__((ext_vector_type(8)))  float    v8f;
typedef __attribute__((ext_vector_type(4)))  float    f4;

#define BB    8
#define NN    1024
#define DINC  256
#define OUTD  128
#define NH    8
#define HEADD 16
#define DEE   8
#define DGG   128

#if defined(__gfx1250__) && \
    __has_builtin(__builtin_amdgcn_global_load_async_to_lds_b128) && \
    __has_builtin(__builtin_amdgcn_global_load_async_to_lds_b32) && \
    __has_builtin(__builtin_amdgcn_s_wait_asynccnt)
#define USE_ASYNC_LDS 1
#else
#define USE_ASYNC_LDS 0
#endif

#if USE_ASYNC_LDS
typedef int i32x4 __attribute__((vector_size(16)));      // GCC-vector int4 (matches builtin proto)
typedef __attribute__((address_space(1))) i32x4 g_i32x4; // global
typedef __attribute__((address_space(3))) i32x4 l_i32x4; // LDS
typedef __attribute__((address_space(1))) int   g_i32;
typedef __attribute__((address_space(3))) int   l_i32;

__device__ __forceinline__ void async_b128(const void* g, const void* l) {
  // low 32 bits of a generic LDS pointer are the LDS byte offset (ISA aperture rule)
  __builtin_amdgcn_global_load_async_to_lds_b128(
      (g_i32x4*)(unsigned long long)g,
      (l_i32x4*)(unsigned int)(unsigned long long)l, 0, 0);
}
__device__ __forceinline__ void async_b32(const void* g, const void* l) {
  __builtin_amdgcn_global_load_async_to_lds_b32(
      (g_i32*)(unsigned long long)g,
      (l_i32*)(unsigned int)(unsigned long long)l, 0, 0);
}
#endif

// ---------------------------------------------------------------------------
// Kernel 1: per 16-node tile, z = concat(nf, hidden) projections via WMMA.
//   values -> f16 workspace, skip -> d_out (f32), att1/att2 -> f32 workspace.
// ---------------------------------------------------------------------------
__global__ __launch_bounds__(32) void k1_proj(
    const float* __restrict__ hidden, const float* __restrict__ nfeat,
    const float* __restrict__ Wm,  const float* __restrict__ bm,
    const float* __restrict__ Wsk, const float* __restrict__ bsk,
    const float* __restrict__ Wa1, const float* __restrict__ ba1,
    const float* __restrict__ Wa2, const float* __restrict__ ba2,
    _Float16* __restrict__ valh, float* __restrict__ att1w,
    float* __restrict__ att2w, float* __restrict__ outp)
{
  const int blk  = blockIdx.x;           // B * N/16 = 512
  const int b    = blk >> 6;
  const int i0   = (blk & 63) << 4;
  const int lane = threadIdx.x;          // wave32
  const int m    = lane & 15;
  const int kb   = (lane < 16) ? 0 : 8;  // A-fragment K base (ISA 16-bit A 16x32)
  const int base8 = kb;                  // C-fragment row base (ISA C/D layout)
  const int n    = lane & 15;
  const int node = i0 + m;

  // Build A fragments once for all 8 K-chunks (K = 256 = 8 * 32).
  v16h afr[8];
#pragma unroll
  for (int c = 0; c < 8; ++c) {
    const float* src = (c < 4)
        ? (nfeat  + ((size_t)(b * NN + node)) * 128 + c * 32 + kb)
        : (hidden + ((size_t)(b * NN + node)) * 128 + (c - 4) * 32 + kb);
    f4 s0 = *(const f4*)(src);
    f4 s1 = *(const f4*)(src + 4);
    f4 s2 = *(const f4*)(src + 16);
    f4 s3 = *(const f4*)(src + 20);
#pragma unroll
    for (int q = 0; q < 4; ++q) {
      afr[c][q]      = (_Float16)s0[q];
      afr[c][4 + q]  = (_Float16)s1[q];
      afr[c][8 + q]  = (_Float16)s2[q];
      afr[c][12 + q] = (_Float16)s3[q];
    }
  }

  // ---- attention head projections: z @ Wa1, z @ Wa2  (8 cols padded to 16)
  {
    float bias1 = (n < NH) ? ba1[n] : 0.f;
    float bias2 = (n < NH) ? ba2[n] : 0.f;
    v8f acc1, acc2;
#pragma unroll
    for (int r = 0; r < 8; ++r) { acc1[r] = bias1; acc2[r] = bias2; }
#pragma unroll
    for (int c = 0; c < 8; ++c) {
      int kq = c * 32 + kb;              // B layout: lane%16 = N col, elements = K (table order)
      v16h b1, b2;
#pragma unroll
      for (int h = 0; h < 16; ++h) {
        b1[h] = (n < NH) ? (_Float16)Wa1[(size_t)(kq + (h < 8 ? h : h + 8)) * NH + n] : (_Float16)0.f;
        b2[h] = (n < NH) ? (_Float16)Wa2[(size_t)(kq + (h < 8 ? h : h + 8)) * NH + n] : (_Float16)0.f;
      }
      acc1 = __builtin_amdgcn_wmma_f32_16x16x32_f16(false, afr[c], false, b1,
                                                    (short)0, acc1, false, false);
      acc2 = __builtin_amdgcn_wmma_f32_16x16x32_f16(false, afr[c], false, b2,
                                                    (short)0, acc2, false, false);
    }
    if (n < NH) {
#pragma unroll
      for (int r = 0; r < 8; ++r) {
        int row = i0 + base8 + r;
        att1w[(size_t)(b * NN + row) * NH + n] = acc1[r];
        att2w[(size_t)(b * NN + row) * NH + n] = acc2[r];
      }
    }
  }

  // ---- values (-> f16 ws) and skip (-> d_out) projections, 8 N-tiles of 16
  for (int nt = 0; nt < 8; ++nt) {
    const int col = nt * 16 + n;
    const float bv = bm[col], bs = bsk[col];
    v8f accv, accs;
#pragma unroll
    for (int r = 0; r < 8; ++r) { accv[r] = bv; accs[r] = bs; }
    for (int c = 0; c < 8; ++c) {
      int k = c * 32 + kb;
      const float* wv = Wm  + (size_t)k * OUTD + col;
      const float* ws = Wsk + (size_t)k * OUTD + col;
      v16h bwv, bws;
#pragma unroll
      for (int h = 0; h < 16; ++h) {
        bwv[h] = (_Float16)wv[(h < 8 ? h : h + 8) * OUTD];
        bws[h] = (_Float16)ws[(h < 8 ? h : h + 8) * OUTD];
      }
      accv = __builtin_amdgcn_wmma_f32_16x16x32_f16(false, afr[c], false, bwv,
                                                    (short)0, accv, false, false);
      accs = __builtin_amdgcn_wmma_f32_16x16x32_f16(false, afr[c], false, bws,
                                                    (short)0, accs, false, false);
    }
#pragma unroll
    for (int r = 0; r < 8; ++r) {
      int row = i0 + base8 + r;
      valh[(size_t)(b * NN + row) * OUTD + col] = (_Float16)accv[r];
      outp[(size_t)(b * NN + row) * OUTD + col] = accs[r];
    }
  }
}

// ---------------------------------------------------------------------------
// Kernel 2: fused logits + flash softmax + coefs@values.  One block (4 waves,
// 2 heads/wave) per (b, 16-row tile); e/adj/att2 double-buffered in LDS via
// CDNA5 async global->LDS loads (ASYNCcnt) when available.
// ---------------------------------------------------------------------------
__device__ __forceinline__ void flash_step(const v16h& lg, float& mr, float& lr,
                                           v8f& acc, const _Float16* bsrc,
                                           int base8)
{
  float cmax = -1e30f;
#pragma unroll
  for (int idx = 0; idx < 16; ++idx) cmax = fmaxf(cmax, (float)lg[idx]);
  cmax = fmaxf(cmax, __shfl_xor(cmax, 16, 32));     // pair lanes L <-> L+16 share a row
  float mnew = fmaxf(mr, cmax);
  float s = __expf(mr - mnew);
  float psum = 0.f;
  v16h pf;
#pragma unroll
  for (int idx = 0; idx < 16; ++idx) {
    float p = __expf((float)lg[idx] - mnew);
    pf[idx] = (_Float16)p;
    psum += p;
  }
  psum += __shfl_xor(psum, 16, 32);
  lr = lr * s + psum;
  mr = mnew;
  // rescale accumulator: C-frag VGPR r holds row base8+r; scale lives in lane==row
#pragma unroll
  for (int r = 0; r < 8; ++r) acc[r] *= __shfl(s, base8 + r, 32);
  v16h bf;                                          // B-frag: lane%16 = head dim, elements = K (table order)
#pragma unroll
  for (int idx = 0; idx < 16; ++idx) bf[idx] = bsrc[(idx < 8 ? idx : idx + 8) * OUTD];
  acc = __builtin_amdgcn_wmma_f32_16x16x32_f16(false, pf, false, bf,
                                               (short)0, acc, false, false);
}

__global__ __launch_bounds__(128) void k2_attn(
    const float* __restrict__ efeat, const float* __restrict__ adj,
    const float* __restrict__ gfeat,
    const float* __restrict__ Wae, const float* __restrict__ bae,
    const float* __restrict__ Wag, const float* __restrict__ bag,
    const _Float16* __restrict__ valh,
    const float* __restrict__ att1w, const float* __restrict__ att2w,
    float* __restrict__ outp)
{
  __shared__ float e_lds[2][16 * 32 * 8];   // 2 x 16 KB
  __shared__ float adj_lds[2][16 * 32];     // 2 x 2 KB
  __shared__ float a2_lds[2][32 * NH];      // 2 x 1 KB
  __shared__ float cg_lds[NH];

  const int blk  = blockIdx.x;
  const int b    = blk >> 6;
  const int i0   = (blk & 63) << 4;
  const int tid  = threadIdx.x;
  const int lane = tid & 31;
  const int wave = tid >> 5;
  const int m    = lane & 15;
  const int kb   = (lane < 16) ? 0 : 8;
  const int base8 = kb;
  const int h0 = wave * 2, h1 = wave * 2 + 1;

  // per-block graph term: cg[h] = g @ Wag + bag + bae
  if (tid < NH) {
    float s = 0.f;
    for (int d = 0; d < DGG; ++d) s += gfeat[b * DGG + d] * Wag[d * NH + tid];
    cg_lds[tid] = s + bag[tid] + bae[tid];
  }

  // stage one 16x32 chunk (e / adj / att2) into LDS buffer bf
  auto stage = [&](int j0, int bf) {
#if USE_ASYNC_LDS
#pragma unroll
    for (int q = 0; q < 4; ++q) {        // e tile: 512 pairs * 8 floats (2x b128 each)
      int p = tid * 4 + q;
      int r = p >> 5, jj = p & 31;
      const float* src = efeat + ((size_t)(b * NN + i0 + r) * NN + (j0 + jj)) * DEE;
      async_b128(src,     &e_lds[bf][p * 8]);
      async_b128(src + 4, &e_lds[bf][p * 8 + 4]);
    }
#pragma unroll
    for (int q = 0; q < 4; ++q) {        // adj tile: 512 floats
      int p = q * 128 + tid;
      int r = p >> 5, jj = p & 31;
      async_b32(adj + (size_t)(b * NN + i0 + r) * NN + (j0 + jj), &adj_lds[bf][p]);
    }
#pragma unroll
    for (int q = 0; q < 2; ++q) {        // att2 tile: 32 nodes * 8 heads
      int p = q * 128 + tid;
      async_b32(att2w + (size_t)(b * NN + j0) * NH + p, &a2_lds[bf][p]);
    }
#else
#pragma unroll
    for (int q = 0; q < 4; ++q) {
      int p = tid * 4 + q;
      int r = p >> 5, jj = p & 31;
      const float* src = efeat + ((size_t)(b * NN + i0 + r) * NN + (j0 + jj)) * DEE;
      *(f4*)(&e_lds[bf][p * 8])     = *(const f4*)src;
      *(f4*)(&e_lds[bf][p * 8 + 4]) = *(const f4*)(src + 4);
    }
#pragma unroll
    for (int q = 0; q < 4; ++q) {
      int p = q * 128 + tid;
      int r = p >> 5, jj = p & 31;
      adj_lds[bf][p] = adj[(size_t)(b * NN + i0 + r) * NN + (j0 + jj)];
    }
#pragma unroll
    for (int q = 0; q < 2; ++q) {
      int p = q * 128 + tid;
      a2_lds[bf][p] = att2w[(size_t)(b * NN + j0) * NH + p];
    }
#endif
  };

  float w0[8], w1[8];
#pragma unroll
  for (int d = 0; d < 8; ++d) { w0[d] = Wae[d * NH + h0]; w1[d] = Wae[d * NH + h1]; }
  const float a1h0 = att1w[(size_t)(b * NN + i0 + m) * NH + h0];
  const float a1h1 = att1w[(size_t)(b * NN + i0 + m) * NH + h1];

  v8f acc0 = {}, acc1 = {};
  float mr0 = -1e30f, mr1 = -1e30f, lr0 = 0.f, lr1 = 0.f;

#if USE_ASYNC_LDS
  stage(0, 0);                       // prologue: chunk 0 in flight
#endif
  __syncthreads();                   // cg_lds ready
  const float cg0 = cg_lds[h0], cg1 = cg_lds[h1];

  for (int ch = 0; ch < 32; ++ch) {
    const int cb = ch & 1;
    __syncthreads();                 // all waves done reading buffer (ch+1)&1
#if USE_ASYNC_LDS
    if (ch + 1 < 32) {
      stage((ch + 1) * 32, (ch + 1) & 1);   // prefetch next chunk (14 async ops)
      __builtin_amdgcn_s_wait_asynccnt(14); // current buffer's loads complete
    } else {
      __builtin_amdgcn_s_wait_asynccnt(0);
    }
#else
    stage(ch * 32, cb);
#endif
    __syncthreads();                 // every wave's partition of buffer cb visible

    const int j0 = ch * 32;
    // logits for this wave's two heads, emitted directly in WMMA A-fragment layout
    v16h lg0, lg1;
#pragma unroll
    for (int idx = 0; idx < 16; ++idx) {
      int k = kb + (idx < 8 ? idx : idx + 8);      // ISA A-layout K mapping
      const float* e8 = &e_lds[cb][(m * 32 + k) * 8];
      float d0 = 0.f, d1 = 0.f;
#pragma unroll
      for (int d = 0; d < 8; ++d) { float e = e8[d]; d0 += e * w0[d]; d1 += e * w1[d]; }
      float bias = (adj_lds[cb][m * 32 + k] - 1.0f) * 1e9f;
      float x0 = a1h0 + a2_lds[cb][k * NH + h0] + cg0 + d0;
      float x1 = a1h1 + a2_lds[cb][k * NH + h1] + cg1 + d1;
      x0 = (x0 > 0.f ? x0 : 0.01f * x0) + bias;    // leaky_relu then mask bias
      x1 = (x1 > 0.f ? x1 : 0.01f * x1) + bias;
      x0 = fmaxf(x0, -60000.f);                    // keep finite in f16
      x1 = fmaxf(x1, -60000.f);
      lg0[idx] = (_Float16)x0;
      lg1[idx] = (_Float16)x1;
    }

    const _Float16* bbase = valh + (size_t)(b * NN + j0 + kb) * OUTD + m;
    flash_step(lg0, mr0, lr0, acc0, bbase + h0 * HEADD, base8);
    flash_step(lg1, mr1, lr1, acc1, bbase + h1 * HEADD, base8);
  }

  // finalize: divide by softmax denominator, add skip residual already in d_out
#pragma unroll
  for (int r = 0; r < 8; ++r) {
    float l0 = __shfl(lr0, base8 + r, 32);
    float l1 = __shfl(lr1, base8 + r, 32);
    int row = i0 + base8 + r;
    size_t o0 = (size_t)(b * NN + row) * OUTD + h0 * HEADD + m;
    size_t o1 = (size_t)(b * NN + row) * OUTD + h1 * HEADD + m;
    outp[o0] += acc0[r] / l0;
    outp[o1] += acc1[r] / l1;
  }
}

// ---------------------------------------------------------------------------
extern "C" void kernel_launch(void* const* d_in, const int* in_sizes, int n_in,
                              void* d_out, int out_size, void* d_ws, size_t ws_size,
                              hipStream_t stream) {
  const float* hidden = (const float*)d_in[0];
  const float* nfeat  = (const float*)d_in[1];
  const float* efeat  = (const float*)d_in[2];
  const float* gfeat  = (const float*)d_in[3];
  const float* adj    = (const float*)d_in[4];
  const float* Wm     = (const float*)d_in[5];
  const float* bm     = (const float*)d_in[6];
  const float* Wsk    = (const float*)d_in[7];
  const float* bsk    = (const float*)d_in[8];
  const float* Wa1    = (const float*)d_in[9];
  const float* ba1    = (const float*)d_in[10];
  const float* Wa2    = (const float*)d_in[11];
  const float* ba2    = (const float*)d_in[12];
  const float* Wae    = (const float*)d_in[13];
  const float* bae    = (const float*)d_in[14];
  const float* Wag    = (const float*)d_in[15];
  const float* bag    = (const float*)d_in[16];
  float* outp = (float*)d_out;

  char* ws = (char*)d_ws;
  _Float16* valh = (_Float16*)ws;                              // 8*1024*128 f16 = 2 MB
  float* att1w = (float*)(ws + (size_t)2 * 1024 * 1024);       // 256 KB
  float* att2w = (float*)(ws + (size_t)2 * 1024 * 1024 + 256 * 1024);

  const int tiles = BB * (NN / 16);  // 512
  k1_proj<<<dim3(tiles), dim3(32), 0, stream>>>(
      hidden, nfeat, Wm, bm, Wsk, bsk, Wa1, ba1, Wa2, ba2,
      valh, att1w, att2w, outp);
  k2_attn<<<dim3(tiles), dim3(128), 0, stream>>>(
      efeat, adj, gfeat, Wae, bae, Wag, bag, valh, att1w, att2w, outp);
}
